// MLA_85907935855048
// MI455X (gfx1250) — matrix operand order repaired and checked by reading, not compiled, not run on hardware
//
#include <hip/hip_runtime.h>
#include <hip/hip_bf16.h>
#include <math.h>

// ---------------------------------------------------------------- constants
static constexpr int kDIM      = 2048;
static constexpr int kHEADS    = 16;
static constexpr int kHD       = 128;     // head_dim
static constexpr int kKVC      = 512;
static constexpr int kQC       = 1024;
static constexpr int kB        = 2;
static constexpr int kS        = 2048;
static constexpr int kROWS     = kB * kS; // 4096

#define USE_ASYNC_LDS 1

typedef __attribute__((ext_vector_type(16))) __bf16        v16bf;
typedef __attribute__((ext_vector_type(8)))  float         v8f;
typedef __attribute__((ext_vector_type(4)))  unsigned int  v4u;

// Fragment: 8 dwords = two 16B halves (matches A/B VGPR layouts, ISA 7.12.2)
union FragB16 { v16bf v; v4u q[2]; unsigned int u[8]; };

// ---------------------------------------------------------------- helpers
__device__ __forceinline__ float bf2f(unsigned short h) {
    unsigned int u = ((unsigned int)h) << 16;
    return __uint_as_float(u);
}
__device__ __forceinline__ unsigned short f2bf(float f) {
    unsigned int u = __float_as_uint(f);
    unsigned int r = (u + 0x7FFFu + ((u >> 16) & 1u)) >> 16;   // RNE
    return (unsigned short)r;
}
__device__ __forceinline__ v8f wmma_bf16(const FragB16& a, const FragB16& b, v8f c) {
    return __builtin_amdgcn_wmma_f32_16x16x32_bf16(
        false, a.v, false, b.v, (short)0, c, false, false);
}
// A-fragment (16x32, 16-bit): lane row = lane%16; K halves at base and 16+base,
// base = (lane/16)*8; each half is 8 contiguous elems = one 16B load.
__device__ __forceinline__ void load_a_frag(FragB16& f, const unsigned short* p, int half) {
    const int base = half * 8;
    f.q[0] = *(const v4u*)(p + base);
    f.q[1] = *(const v4u*)(p + 16 + base);
}
// B-fragment (32x16, 16-bit): lane col = lane%16; K halves at base and 16+base,
// base = (lane/16)*8, matching the A layout = two 16B loads.
__device__ __forceinline__ void load_b_frag(FragB16& f, const unsigned short* p, int half) {
    const unsigned short* bp = p + half * 8;
    f.q[0] = *(const v4u*)(bp);
    f.q[1] = *(const v4u*)(bp + 16);
}

// ---------------------------------------------------------------- fp32 -> bf16
__global__ void cvt_f32_bf16_kernel(const float* __restrict__ in,
                                    unsigned short* __restrict__ out, int n) {
    for (int i = blockIdx.x * blockDim.x + threadIdx.x; i < n;
         i += gridDim.x * blockDim.x)
        out[i] = f2bf(in[i]);
}

// ---------------------------------------------------------------- GEMM bf16
// C[M,N] = A[M,K] @ B[K,N]; A,B bf16 row-major; C -> bf16 or f32.
// Block = 256 threads = 8 waves; block tile 128x128; wave tile 32x64.
// B tile staged 64-deep, transposed, row stride 72 elems (16B aligned,
// 36-dword stride -> conflict-free 4-bank groups for 16 lanes).
static constexpr int kBT_STRIDE = 72;     // 64 + 8 pad elems (36 dwords)

__global__ __launch_bounds__(256, 1)
void gemm_bf16_kernel(const unsigned short* __restrict__ A,
                      const unsigned short* __restrict__ Bm,
                      unsigned short* __restrict__ Cb,
                      float* __restrict__ Cf,
                      int M, int N, int K, int storeF32) {
    __shared__ unsigned short Bt[128 * kBT_STRIDE];
    const int tid  = threadIdx.x;
    const int lane = tid & 31;
    const int w    = tid >> 5;
    const int wr   = w >> 1;        // 0..3
    const int wc   = w & 1;         // 0..1
    const int half = lane >> 4;
    const int nn   = lane & 15;
    const int rowBase = blockIdx.y * 128;
    const int colBase = blockIdx.x * 128;

    v8f acc[2][4];
#pragma unroll
    for (int rt = 0; rt < 2; ++rt)
#pragma unroll
        for (int ct = 0; ct < 4; ++ct)
#pragma unroll
            for (int e = 0; e < 8; ++e) acc[rt][ct][e] = 0.0f;

    for (int kt = 0; kt < K; kt += 64) {
        __syncthreads();
        // stage B[k, n] -> Bt[n, k]  (64 x 128 tile)
        for (int idx = tid; idx < 8192; idx += 256) {
            int k = idx >> 7, n = idx & 127;
            Bt[n * kBT_STRIDE + k] = Bm[(size_t)(kt + k) * N + colBase + n];
        }
        if (kt + 64 < K)
            __builtin_prefetch(Bm + (size_t)(kt + 64) * N + colBase + tid, 0, 0);
        __syncthreads();

#pragma unroll
        for (int cc = 0; cc < 2; ++cc) {            // two K-chunks of 32
            FragB16 afrag[2];
#pragma unroll
            for (int rt = 0; rt < 2; ++rt) {
                int row = rowBase + wr * 32 + rt * 16 + nn;
                load_a_frag(afrag[rt], A + (size_t)row * K + kt + cc * 32, half);
            }
#pragma unroll
            for (int ct = 0; ct < 4; ++ct) {
                FragB16 bfrag;
                load_b_frag(bfrag,
                            &Bt[(wc * 64 + ct * 16 + nn) * kBT_STRIDE + cc * 32],
                            half);
#pragma unroll
                for (int rt = 0; rt < 2; ++rt)
                    acc[rt][ct] = wmma_bf16(afrag[rt], bfrag, acc[rt][ct]);
            }
        }
    }

    // store (C layout: VGPR r -> row r + 8*half, col = lane%16)
#pragma unroll
    for (int rt = 0; rt < 2; ++rt)
#pragma unroll
        for (int ct = 0; ct < 4; ++ct)
#pragma unroll
            for (int r = 0; r < 8; ++r) {
                int m   = rowBase + wr * 32 + rt * 16 + r + 8 * half;
                int col = colBase + wc * 64 + ct * 16 + nn;
                float val = acc[rt][ct][r];
                if (storeF32) Cf[(size_t)m * N + col] = val;
                else          Cb[(size_t)m * N + col] = f2bf(val);
            }
}

// ---------------------------------------------------------------- RoPE (in place)
__global__ void rope_kernel(unsigned short* __restrict__ xr, int rows, int s_len) {
    int t = blockIdx.x * blockDim.x + threadIdx.x;
    if (t >= rows * 64) return;
    int row = t >> 6, i = t & 63;
    int pos = row % s_len;
    float inv = expf(-(2.0f * (float)i / 128.0f) * logf(10000.0f));
    float ang = (float)pos * inv;
    float c = cosf(ang), s = sinf(ang);
    unsigned short* p = xr + (size_t)row * kHD + 2 * i;
    float x0 = bf2f(p[0]), x1 = bf2f(p[1]);
    p[0] = f2bf(x0 * c - x1 * s);
    p[1] = f2bf(x1 * c + x0 * s);
}

// ---------------------------------------------------------------- flash attention
// q = [q_c | q_rope] (256), k = [k_c | k_rope] (256), v = v_c (128).
// Block: 256 thr = 8 waves, each wave owns 16 q rows; 64-key tiles in LDS.
// All row strides multiples of 8 elems so fragment reads are ds_load_b128.
static constexpr int kKT_STRIDE = 264;   // 256 + 8 pad (132 dwords)
static constexpr int kVT_STRIDE = 72;    // 64 + 8 pad  (36 dwords, conflict-free)
static constexpr int kP_STRIDE  = 72;
static constexpr int kLDS_ELEMS = 64 * kKT_STRIDE + 128 * kVT_STRIDE + 8 * 16 * kP_STRIDE;

__global__ __launch_bounds__(256, 1)
void mla_flash_attn_kernel(const unsigned short* __restrict__ qc,
                           const unsigned short* __restrict__ qr,
                           const unsigned short* __restrict__ kc,
                           const unsigned short* __restrict__ kr,
                           const unsigned short* __restrict__ vc,
                           unsigned short* __restrict__ attn_out) {
    extern __shared__ unsigned short smem[];
    unsigned short* Kt = smem;                       // [64][264]  K tile, key-major
    unsigned short* Vt = Kt + 64 * kKT_STRIDE;       // [128][72]  V tile transposed
    unsigned short* Pb = Vt + 128 * kVT_STRIDE;      // per-wave [16][72]

    const float scale = 0.0625f;   // 1/sqrt(256)
    const int tid  = threadIdx.x;
    const int lane = tid & 31;
    const int w    = tid >> 5;
    const int half = lane >> 4;
    const int nn   = lane & 15;
    const int b = blockIdx.z, h = blockIdx.y;
    const int qBase = blockIdx.x * 128 + w * 16;
    const size_t rowQ0 = (size_t)b * kS + qBase;
    unsigned short* Pw = Pb + w * (16 * kP_STRIDE);

    // ---- resident q fragments: 8 chunks x 32 dims (0..127 = q_c, 128..255 = rope)
    FragB16 qf[8];
#pragma unroll
    for (int c = 0; c < 8; ++c) {
        size_t row = rowQ0 + nn;
        const unsigned short* qp = (c < 4)
            ? qc + row * kDIM + (size_t)h * kHD + c * 32
            : qr + row * kHD + (c - 4) * 32;
        load_a_frag(qf[c], qp, half);
    }

    v8f O[8];
    float m_run[8], l_run[8];
#pragma unroll
    for (int ot = 0; ot < 8; ++ot)
#pragma unroll
        for (int e = 0; e < 8; ++e) O[ot][e] = 0.0f;
#pragma unroll
    for (int r = 0; r < 8; ++r) { m_run[r] = -1e30f; l_run[r] = 0.0f; }

    for (int kb = 0; kb < kS; kb += 64) {
        __syncthreads();

        // ---- stage K tile: Kt[key][0..255] = [k_c | k_rope]
#if USE_ASYNC_LDS
        // contiguous 16B chunks: async DMA global -> LDS, no VGPR round-trip
        for (int ci = tid; ci < 64 * 32; ci += 256) {
            int kj = ci >> 5, c8 = ci & 31;
            int d0 = c8 * 8;                         // 8 elems = 16 bytes
            size_t rk = (size_t)b * kS + kb + kj;
            const unsigned short* gp = (d0 < 128)
                ? kc + rk * kDIM + (size_t)h * kHD + d0
                : kr + rk * kHD + (d0 - 128);
            unsigned int ldsoff = (unsigned int)
                ((const char*)&Kt[kj * kKT_STRIDE + d0] - (const char*)smem);
            unsigned long long ga = (unsigned long long)(uintptr_t)gp;
            asm volatile("global_load_async_to_lds_b128 %0, %1, off"
                         :: "v"(ldsoff), "v"(ga) : "memory");
        }
#else
        for (int idx = tid; idx < 64 * 256; idx += 256) {
            int d = idx & 255, kj = idx >> 8;
            size_t rk = (size_t)b * kS + kb + kj;
            unsigned short val = (d < 128)
                ? kc[rk * kDIM + (size_t)h * kHD + d]
                : kr[rk * kHD + (d - 128)];
            Kt[kj * kKT_STRIDE + d] = val;
        }
#endif
        // ---- stage V tile transposed: Vt[d][key] (overlaps with async K copy)
        for (int idx = tid; idx < 64 * 128; idx += 256) {
            int d = idx & 127, kj = idx >> 7;
            size_t rk = (size_t)b * kS + kb + kj;
            Vt[d * kVT_STRIDE + kj] = vc[rk * kDIM + (size_t)h * kHD + d];
        }
#if USE_ASYNC_LDS
        asm volatile("s_wait_asynccnt 0x0" ::: "memory");
#endif
        __syncthreads();

        // ---- scores: 4 key-column tiles x 8 K-chunks of WMMA
        v8f sacc[4];
#pragma unroll
        for (int ct = 0; ct < 4; ++ct) {
#pragma unroll
            for (int e = 0; e < 8; ++e) sacc[ct][e] = 0.0f;
#pragma unroll
            for (int c = 0; c < 8; ++c) {
                FragB16 kf;  // B = K^T: N = key (lane%16), Kdim = d
                load_b_frag(kf, &Kt[(ct * 16 + nn) * kKT_STRIDE + c * 32], half);
                sacc[ct] = wmma_bf16(qf[c], kf, sacc[ct]);
            }
        }

        // ---- online softmax (rows live in VGPR index r + 8*half; cols in lanes)
        float mnew[8], alpha[8];
#pragma unroll
        for (int r = 0; r < 8; ++r) {
            float t = fmaxf(fmaxf(sacc[0][r], sacc[1][r]),
                            fmaxf(sacc[2][r], sacc[3][r]));
#pragma unroll
            for (int msk = 1; msk < 16; msk <<= 1)
                t = fmaxf(t, __shfl_xor(t, msk, 32));
            t *= scale;
            mnew[r]  = fmaxf(m_run[r], t);
            alpha[r] = __expf(m_run[r] - mnew[r]);
        }
#pragma unroll
        for (int r = 0; r < 8; ++r) {
            int m_row = r + 8 * half;
            float rs = 0.0f;
#pragma unroll
            for (int ct = 0; ct < 4; ++ct) {
                float p = __expf(sacc[ct][r] * scale - mnew[r]);
                rs += p;
                Pw[m_row * kP_STRIDE + ct * 16 + nn] = f2bf(p);
            }
#pragma unroll
            for (int msk = 1; msk < 16; msk <<= 1)
                rs += __shfl_xor(rs, msk, 32);
            l_run[r] = l_run[r] * alpha[r] + rs;
            m_run[r] = mnew[r];
        }
#pragma unroll
        for (int ot = 0; ot < 8; ++ot)
#pragma unroll
            for (int r = 0; r < 8; ++r) O[ot][r] *= alpha[r];

        // wave-private LDS bounce: make P stores visible before A-frag reads
        asm volatile("s_wait_dscnt 0x0" ::: "memory");

        // ---- O += P @ V  (2 key chunks of 32 x 8 d-column tiles)
#pragma unroll
        for (int c2 = 0; c2 < 2; ++c2) {
            FragB16 pf;  // A = P: row = lane%16, Kdim = key
            load_a_frag(pf, Pw + nn * kP_STRIDE + c2 * 32, half);
#pragma unroll
            for (int ot = 0; ot < 8; ++ot) {
                FragB16 vf;  // B = V: N = d (lane%16), Kdim = key
                load_b_frag(vf, &Vt[(ot * 16 + nn) * kVT_STRIDE + c2 * 32], half);
                O[ot] = wmma_bf16(pf, vf, O[ot]);
            }
        }
    }

    // ---- finalize: O /= l, store bf16 into attn_out [B*S, H*HD]
#pragma unroll
    for (int r = 0; r < 8; ++r) l_run[r] = 1.0f / l_run[r];
#pragma unroll
    for (int ot = 0; ot < 8; ++ot)
#pragma unroll
        for (int r = 0; r < 8; ++r) {
            int m_row = r + 8 * half;
            size_t row = rowQ0 + m_row;
            attn_out[row * kDIM + (size_t)h * kHD + ot * 16 + nn] =
                f2bf(O[ot][r] * l_run[r]);
        }
}

// ---------------------------------------------------------------- host
extern "C" void kernel_launch(void* const* d_in, const int* in_sizes, int n_in,
                              void* d_out, int out_size, void* d_ws, size_t ws_size,
                              hipStream_t stream) {
    unsigned short* ws = (unsigned short*)d_ws;
    size_t off = 0;
    auto alloc = [&](size_t n) { size_t o = off; off += (n + 63) & ~(size_t)63; return o; };

    // bf16 copies of inputs
    const size_t oX    = alloc((size_t)kROWS * kDIM);
    const size_t oWkvd = alloc((size_t)kDIM * kKVC);
    const size_t oWupk = alloc((size_t)kKVC * kDIM);
    const size_t oWupv = alloc((size_t)kKVC * kDIM);
    const size_t oWqd  = alloc((size_t)kDIM * kQC);
    const size_t oWqu  = alloc((size_t)kQC * kDIM);
    const size_t oWkr  = alloc((size_t)kDIM * kHD);
    const size_t oWqr  = alloc((size_t)kDIM * kHD);
    const size_t oWout = alloc((size_t)kDIM * kDIM);
    // intermediates
    const size_t oKvLat = alloc((size_t)kROWS * kKVC);
    const size_t oKc    = alloc((size_t)kROWS * kDIM);
    const size_t oVc    = alloc((size_t)kROWS * kDIM);
    const size_t oQLat  = alloc((size_t)kROWS * kQC);
    const size_t oQc    = alloc((size_t)kROWS * kDIM);
    const size_t oKr    = alloc((size_t)kROWS * kHD);
    const size_t oQr    = alloc((size_t)kROWS * kHD);
    const size_t oAttn  = alloc((size_t)kROWS * kDIM);
    (void)ws_size;

    // 1) convert all fp32 inputs to bf16
    const size_t cvtOff[9] = {oX, oWkvd, oWupk, oWupv, oWqd, oWqu, oWkr, oWqr, oWout};
    for (int i = 0; i < 9 && i < n_in; ++i) {
        int n = in_sizes[i];
        int blocks = (n + 255) / 256;
        if (blocks > 4096) blocks = 4096;
        cvt_f32_bf16_kernel<<<blocks, 256, 0, stream>>>(
            (const float*)d_in[i], ws + cvtOff[i], n);
    }

    auto gemm_bf = [&](size_t a, size_t bm, size_t c, int M, int N, int K) {
        gemm_bf16_kernel<<<dim3(N / 128, M / 128), 256, 0, stream>>>(
            ws + a, ws + bm, ws + c, nullptr, M, N, K, 0);
    };

    // 2) projections
    gemm_bf(oX,     oWkvd, oKvLat, kROWS, kKVC, kDIM);   // kv_latent
    gemm_bf(oKvLat, oWupk, oKc,    kROWS, kDIM, kKVC);   // k_c
    gemm_bf(oKvLat, oWupv, oVc,    kROWS, kDIM, kKVC);   // v_c
    gemm_bf(oX,     oWqd,  oQLat,  kROWS, kQC,  kDIM);   // q_latent
    gemm_bf(oQLat,  oWqu,  oQc,    kROWS, kDIM, kQC);    // q_c
    gemm_bf(oX,     oWkr,  oKr,    kROWS, kHD,  kDIM);   // k_rope proj
    gemm_bf(oX,     oWqr,  oQr,    kROWS, kHD,  kDIM);   // q_rope proj

    // 3) RoPE (in place; head-independent since broadcast precedes rotation)
    {
        int n = kROWS * 64, blocks = (n + 255) / 256;
        rope_kernel<<<blocks, 256, 0, stream>>>(ws + oKr, kROWS, kS);
        rope_kernel<<<blocks, 256, 0, stream>>>(ws + oQr, kROWS, kS);
    }

    // 4) flash attention
    {
        const int ldsBytes = kLDS_ELEMS * 2;  // ~69 KB, needs opt-in
        (void)hipFuncSetAttribute((const void*)mla_flash_attn_kernel,
                                  hipFuncAttributeMaxDynamicSharedMemorySize,
                                  ldsBytes);
        mla_flash_attn_kernel<<<dim3(kS / 128, kHEADS, kB), 256, ldsBytes, stream>>>(
            ws + oQc, ws + oQr, ws + oKc, ws + oKr, ws + oVc, ws + oAttn);
    }

    // 5) output projection -> f32 d_out
    gemm_bf16_kernel<<<dim3(kDIM / 128, kROWS / 128), 256, 0, stream>>>(
        ws + oAttn, ws + oWout, nullptr, (float*)d_out, kROWS, kDIM, kDIM, 1);
    (void)out_size;
}
